// NNConv_27685359190281
// MI455X (gfx1250) — matrix operand order repaired and checked by reading, not compiled, not run on hardware
//
#include <hip/hip_runtime.h>

#define N_NODES 50000
#define N_EDGES 100000
#define N_GRAPH 256
#define EPS 1e-5f

typedef __attribute__((ext_vector_type(16))) _Float16 v16h;
typedef __attribute__((ext_vector_type(8)))  _Float16 v8h;
typedef __attribute__((ext_vector_type(8)))  float    v8f;

// ---------------------------------------------------------------- utilities
__device__ inline void atomicMaxF(float* addr, float v) {
    // sign-split trick: positive floats compare as ints, negative as reversed uints
    if (!(__float_as_uint(v) >> 31)) atomicMax((int*)addr, __float_as_int(v));
    else                             atomicMin((unsigned int*)addr, __float_as_uint(v));
}

// load a 32-feature row as two packed v16h
__device__ inline void load_row32(const float* __restrict__ p, v16h& lo, v16h& hi) {
    const float4* xp = (const float4*)p;
    #pragma unroll
    for (int qd = 0; qd < 4; ++qd) {
        float4 v = xp[qd];
        lo[qd*4+0] = (_Float16)v.x; lo[qd*4+1] = (_Float16)v.y;
        lo[qd*4+2] = (_Float16)v.z; lo[qd*4+3] = (_Float16)v.w;
    }
    #pragma unroll
    for (int qd = 0; qd < 4; ++qd) {
        float4 v = xp[4 + qd];
        hi[qd*4+0] = (_Float16)v.x; hi[qd*4+1] = (_Float16)v.y;
        hi[qd*4+2] = (_Float16)v.z; hi[qd*4+3] = (_Float16)v.w;
    }
}
__device__ inline void load_row32(const _Float16* __restrict__ p, v16h& lo, v16h& hi) {
    lo = *(const v16h*)p;
    hi = *(const v16h*)(p + 16);
}

// ---------------------------------------------------------------- init
__global__ void init_kernel(float* agg1, float* agg2, float* gmax, float* gsum, float* gcnt) {
    int i = blockIdx.x * blockDim.x + threadIdx.x;
    if (i < N_NODES * 64) agg2[i] = 0.0f;
    if (i < N_NODES * 32) agg1[i] = 0.0f;
    if (i < N_GRAPH * 64) { gmax[i] = __int_as_float(0xff800000); gsum[i] = 0.0f; }
    if (i < N_GRAPH)      gcnt[i] = 0.0f;
}

// ---------------------------------------------------------------- edge MLP (both layers)
__global__ void edge_mlp_kernel(const float* __restrict__ ea,
                                const float* __restrict__ w1a, const float* __restrict__ b1a,
                                const float* __restrict__ w1b, const float* __restrict__ b1b,
                                _Float16* __restrict__ h1e, _Float16* __restrict__ h2e) {
    int e = blockIdx.x * blockDim.x + threadIdx.x;
    if (e >= N_EDGES) return;
    float a[8];
    #pragma unroll
    for (int d = 0; d < 8; ++d) a[d] = ea[(size_t)e * 8 + d];
    for (int u = 0; u < 32; ++u) {
        float s1 = b1a[u], s2 = b1b[u];
        #pragma unroll
        for (int d = 0; d < 8; ++d) { s1 += a[d] * w1a[d*32 + u]; s2 += a[d] * w1b[d*32 + u]; }
        h1e[(size_t)e*32 + u] = (_Float16)fmaxf(s1, 0.0f);
        h2e[(size_t)e*32 + u] = (_Float16)fmaxf(s2, 0.0f);
    }
}

// ---------------------------------------------------------------- reorder w2 -> B fragments
// out[((c*2+nt)*32 + lane)*16 + j] = w2[(lane>>4)*16 + j, c*dout + ch_base + nt*16 + (lane&15)]
__global__ void reorder_b_kernel(const float* __restrict__ w2, _Float16* __restrict__ out,
                                 int dout, int ch_base) {
    int idx = blockIdx.x * blockDim.x + threadIdx.x;
    if (idx >= 32 * 2 * 32 * 16) return;
    int j  = idx & 15;
    int l  = (idx >> 4) & 31;
    int nt = (idx >> 9) & 1;
    int c  = idx >> 10;
    int krow = (l >> 4) * 16 + j;
    int col  = c * dout + ch_base + nt * 16 + (l & 15);
    out[idx] = (_Float16)w2[(size_t)krow * (32 * dout) + col];
}

// ---------------------------------------------------------------- q = in @ reshape(b2, [32,dout])
template<typename TIn>
__global__ void xw_kernel(const TIn* __restrict__ in, const float* __restrict__ w,
                          float* __restrict__ out, int dout) {
    int n = blockIdx.x * blockDim.x + threadIdx.x;
    if (n >= N_NODES) return;
    float xr[32];
    #pragma unroll
    for (int i = 0; i < 32; ++i) xr[i] = (float)in[(size_t)n*32 + i];
    for (int o = 0; o < dout; ++o) {
        float s = 0.0f;
        #pragma unroll
        for (int i = 0; i < 32; ++i) s += xr[i] * w[i*dout + o];
        out[(size_t)n * dout + o] = s;
    }
}

// ---------------------------------------------------------------- fused edge GEMM (WMMA)
// msg[e, ch_base + 0..31] = (x[src[e]] (x) h_e[e]) @ T ,   T staged in fragment order.
// Two 16-edge tiles per wave iteration: each B fragment feeds 2 WMMAs.
template<typename TIn>
__global__ __launch_bounds__(256) void edge_gemm_kernel(
    const TIn*      __restrict__ xin,   // [N,32] node features (f32 layer1, f16 layer2)
    const _Float16* __restrict__ he,    // [E,32] edge hidden (f16)
    const _Float16* __restrict__ bstg,  // [32*2*32*16] staged B fragments
    const int*      __restrict__ ei,    // [2,E] edge_index
    const float*    __restrict__ q,     // [N, dout_total] b2 term per source node
    float*          __restrict__ agg,   // [N, dout_total] scatter target
    int dout_total, int ch_base, int npairs)
{
    __shared__ __align__(32) _Float16 Blds[32 * 2 * 32 * 16];  // 64 KB
    {
        const uint4* s = (const uint4*)bstg;
        uint4* d = (uint4*)Blds;
        for (int i = threadIdx.x; i < (32*2*32*16)/8; i += blockDim.x) d[i] = s[i];
    }
    __syncthreads();

    const int lane  = threadIdx.x & 31;
    const int m     = lane & 15;   // A-matrix row (edge within tile) for this lane
    const int hl    = lane >> 4;   // lane-half selects K sub-range
    const int nwave = (gridDim.x * blockDim.x) >> 5;
    int wave = (blockIdx.x * blockDim.x + threadIdx.x) >> 5;

    const _Float16* lbase = Blds + (lane << 4);
    int zoff = 0;   // stays 0, but opaque to the compiler (defeats LICM of LDS loads)

    for (int pair = wave; pair < npairs; pair += nwave) {
        asm volatile("" : "+v"(zoff));
        const _Float16* lb = lbase + zoff;

        const int t0 = pair * 2, t1 = pair * 2 + 1;
        const int e0 = t0 * 16 + m, e1 = t1 * 16 + m;
        const int s0 = ei[e0], s1 = ei[e1];

        // ---- x rows (packed halves)
        v16h x0l, x0h, x1l, x1h;
        load_row32(xin + (size_t)s0 * 32, x0l, x0h);
        load_row32(xin + (size_t)s1 * 32, x1l, x1h);

        // ---- h fragments: lane l holds K = hl*16+{0..15} (same k order as staged B)
        v8h a0 = *(const v8h*)(he + (size_t)e0*32 + hl*16);
        v8h b0_ = *(const v8h*)(he + (size_t)e0*32 + hl*16 + 8);
        v8h a1 = *(const v8h*)(he + (size_t)e1*32 + hl*16);
        v8h b1_ = *(const v8h*)(he + (size_t)e1*32 + hl*16 + 8);
        v16h hf0, hf1;
        #pragma unroll
        for (int j = 0; j < 8; ++j) {
            hf0[j] = a0[j]; hf0[8 + j] = b0_[j];
            hf1[j] = a1[j]; hf1[8 + j] = b1_[j];
        }

        v8f acc00 = {}, acc01 = {}, acc10 = {}, acc11 = {};
        #pragma unroll     // full unroll: xs extraction and LDS offsets are compile-time
        for (int c = 0; c < 32; ++c) {
            const _Float16 xs0 = (c < 16) ? x0l[c] : x0h[c - 16];
            const _Float16 xs1 = (c < 16) ? x1l[c] : x1h[c - 16];
            v16h sp0, sp1;
            #pragma unroll
            for (int j = 0; j < 16; ++j) { sp0[j] = xs0; sp1[j] = xs1; }
            v16h fa0 = hf0 * sp0;            // A fragment tile0 = x_scalar * h fragment
            v16h fa1 = hf1 * sp1;            // A fragment tile1
            v16h bb0 = *(const v16h*)(lb + (c*2 + 0)*512);
            v16h bb1 = *(const v16h*)(lb + (c*2 + 1)*512);
            acc00 = __builtin_amdgcn_wmma_f32_16x16x32_f16(false, fa0, false, bb0, (short)0, acc00, false, false);
            acc10 = __builtin_amdgcn_wmma_f32_16x16x32_f16(false, fa1, false, bb0, (short)0, acc10, false, false);
            acc01 = __builtin_amdgcn_wmma_f32_16x16x32_f16(false, fa0, false, bb1, (short)0, acc01, false, false);
            acc11 = __builtin_amdgcn_wmma_f32_16x16x32_f16(false, fa1, false, bb1, (short)0, acc11, false, false);
        }

        // ---- scatter: C layout lane l, reg r -> edge row r + 8*hl, channel lane&15
        const int n0 = ch_base + (lane & 15);
        #pragma unroll
        for (int r = 0; r < 8; ++r) {
            const int er = t0*16 + r + 8*hl;
            const int dn = ei[N_EDGES + er];
            const int sn = ei[er];
            const size_t qb = (size_t)sn * dout_total;
            atomicAdd(agg + (size_t)dn * dout_total + n0,      acc00[r] + q[qb + n0]);
            atomicAdd(agg + (size_t)dn * dout_total + n0 + 16, acc01[r] + q[qb + n0 + 16]);
        }
        #pragma unroll
        for (int r = 0; r < 8; ++r) {
            const int er = t1*16 + r + 8*hl;
            const int dn = ei[N_EDGES + er];
            const int sn = ei[er];
            const size_t qb = (size_t)sn * dout_total;
            atomicAdd(agg + (size_t)dn * dout_total + n0,      acc10[r] + q[qb + n0]);
            atomicAdd(agg + (size_t)dn * dout_total + n0 + 16, acc11[r] + q[qb + n0 + 16]);
        }
    }
}

// ---------------------------------------------------------------- node update layer 1 (+relu, ->f16)
__global__ void node1_kernel(const float* __restrict__ x, const float* __restrict__ root,
                             const float* __restrict__ bias, const float* __restrict__ agg,
                             _Float16* __restrict__ hout) {
    int n = blockIdx.x * blockDim.x + threadIdx.x;
    if (n >= N_NODES) return;
    float xr[32];
    #pragma unroll
    for (int i = 0; i < 32; ++i) xr[i] = x[(size_t)n*32 + i];
    for (int o = 0; o < 32; ++o) {
        float sv = bias[o] + agg[(size_t)n*32 + o];
        #pragma unroll
        for (int i = 0; i < 32; ++i) sv += xr[i] * root[i*32 + o];
        hout[(size_t)n*32 + o] = (_Float16)fmaxf(sv, 0.0f);
    }
}

// ---------------------------------------------------------------- node update layer 2 + pooling
__global__ void node2_pool_kernel(const _Float16* __restrict__ h1, const float* __restrict__ root,
                                  const float* __restrict__ bias, const float* __restrict__ agg,
                                  const int* __restrict__ bat,
                                  float* __restrict__ gmax, float* __restrict__ gsum,
                                  float* __restrict__ gcnt) {
    int n = blockIdx.x * blockDim.x + threadIdx.x;
    if (n >= N_NODES) return;
    float hr[32];
    #pragma unroll
    for (int i = 0; i < 32; ++i) hr[i] = (float)h1[(size_t)n*32 + i];
    int g = bat[n];
    atomicAdd(&gcnt[g], 1.0f);
    for (int o = 0; o < 64; ++o) {
        float sv = bias[o] + agg[(size_t)n*64 + o];
        #pragma unroll
        for (int i = 0; i < 32; ++i) sv += hr[i] * root[i*64 + o];
        atomicMaxF(&gmax[g*64 + o], sv);
        atomicAdd(&gsum[g*64 + o], sv);
    }
}

// ---------------------------------------------------------------- MLP head (one thread per graph)
__global__ void head_kernel(const float* __restrict__ gmax, const float* __restrict__ gsum,
                            const float* __restrict__ gcnt,
                            const float* __restrict__ h1w, const float* __restrict__ h1b,
                            const float* __restrict__ bn1g, const float* __restrict__ bn1b,
                            const float* __restrict__ bn1m, const float* __restrict__ bn1v,
                            const float* __restrict__ h2w, const float* __restrict__ h2b,
                            const float* __restrict__ bn2g, const float* __restrict__ bn2b,
                            const float* __restrict__ bn2m, const float* __restrict__ bn2v,
                            const float* __restrict__ h3w, const float* __restrict__ h3b,
                            float* __restrict__ out) {
    int g = blockIdx.x * blockDim.x + threadIdx.x;
    if (g >= N_GRAPH) return;
    float feat[128];
    float cnt = fmaxf(gcnt[g], 1.0f);
    for (int o = 0; o < 64; ++o) feat[o]      = gmax[g*64 + o];
    for (int o = 0; o < 64; ++o) feat[64 + o] = gsum[g*64 + o] / cnt;
    float z1[64];
    for (int o = 0; o < 64; ++o) {
        float s = h1b[o];
        for (int k = 0; k < 128; ++k) s += feat[k] * h1w[k*64 + o];
        s = (s - bn1m[o]) * rsqrtf(bn1v[o] + EPS) * bn1g[o] + bn1b[o];
        z1[o] = fmaxf(s, 0.0f);
    }
    float z2[32];
    for (int o = 0; o < 32; ++o) {
        float s = h2b[o];
        for (int k = 0; k < 64; ++k) s += z1[k] * h2w[k*32 + o];
        s = (s - bn2m[o]) * rsqrtf(bn2v[o] + EPS) * bn2g[o] + bn2b[o];
        z2[o] = fmaxf(s, 0.0f);
    }
    float s = h3b[0];
    for (int k = 0; k < 32; ++k) s += z2[k] * h3w[k];
    out[g] = s;
}

// ---------------------------------------------------------------- launch
extern "C" void kernel_launch(void* const* d_in, const int* in_sizes, int n_in,
                              void* d_out, int out_size, void* d_ws, size_t ws_size,
                              hipStream_t stream) {
    const float* x     = (const float*)d_in[0];
    const int*   ei    = (const int*)  d_in[1];
    const float* ea    = (const float*)d_in[2];
    const int*   bat   = (const int*)  d_in[3];
    const float* e1w1  = (const float*)d_in[4];
    const float* e1b1  = (const float*)d_in[5];
    const float* e1w2  = (const float*)d_in[6];
    const float* e1b2  = (const float*)d_in[7];
    const float* root1 = (const float*)d_in[8];
    const float* bias1 = (const float*)d_in[9];
    const float* e2w1  = (const float*)d_in[10];
    const float* e2b1  = (const float*)d_in[11];
    const float* e2w2  = (const float*)d_in[12];
    const float* e2b2  = (const float*)d_in[13];
    const float* root2 = (const float*)d_in[14];
    const float* bias2 = (const float*)d_in[15];
    const float* h1w = (const float*)d_in[16]; const float* h1b = (const float*)d_in[17];
    const float* bn1g = (const float*)d_in[18]; const float* bn1b = (const float*)d_in[19];
    const float* bn1m = (const float*)d_in[20]; const float* bn1v = (const float*)d_in[21];
    const float* h2w = (const float*)d_in[22]; const float* h2b = (const float*)d_in[23];
    const float* bn2g = (const float*)d_in[24]; const float* bn2b = (const float*)d_in[25];
    const float* bn2m = (const float*)d_in[26]; const float* bn2v = (const float*)d_in[27];
    const float* h3w = (const float*)d_in[28]; const float* h3b = (const float*)d_in[29];
    float* out = (float*)d_out;

    // workspace carve-up (256B aligned)
    unsigned char* w = (unsigned char*)d_ws;
    auto carve = [&](size_t bytes) { void* p = (void*)w; w += (bytes + 255) & ~(size_t)255; return p; };
    float*    agg1 = (float*)   carve((size_t)N_NODES * 32 * 4);
    float*    agg2 = (float*)   carve((size_t)N_NODES * 64 * 4);
    float*    gmax = (float*)   carve((size_t)N_GRAPH * 64 * 4);
    float*    gsum = (float*)   carve((size_t)N_GRAPH * 64 * 4);
    float*    gcnt = (float*)   carve((size_t)N_GRAPH * 4);
    _Float16* h1e  = (_Float16*)carve((size_t)N_EDGES * 32 * 2);
    _Float16* h2e  = (_Float16*)carve((size_t)N_EDGES * 32 * 2);
    _Float16* h1n  = (_Float16*)carve((size_t)N_NODES * 32 * 2);
    float*    q1   = (float*)   carve((size_t)N_NODES * 32 * 4);
    float*    q2   = (float*)   carve((size_t)N_NODES * 64 * 4);
    _Float16* b1s  = (_Float16*)carve((size_t)32768 * 2);
    _Float16* b2sa = (_Float16*)carve((size_t)32768 * 2);
    _Float16* b2sb = (_Float16*)carve((size_t)32768 * 2);

    const int npairs = N_EDGES / 32;   // 3125 pairs of 16-edge tiles
    const int TB = 256;

    init_kernel<<<(N_NODES*64 + TB-1)/TB, TB, 0, stream>>>(agg1, agg2, gmax, gsum, gcnt);
    edge_mlp_kernel<<<(N_EDGES + TB-1)/TB, TB, 0, stream>>>(ea, e1w1, e1b1, e2w1, e2b1, h1e, h2e);
    reorder_b_kernel<<<(32768 + TB-1)/TB, TB, 0, stream>>>(e1w2, b1s,  32, 0);
    reorder_b_kernel<<<(32768 + TB-1)/TB, TB, 0, stream>>>(e2w2, b2sa, 64, 0);
    reorder_b_kernel<<<(32768 + TB-1)/TB, TB, 0, stream>>>(e2w2, b2sb, 64, 32);
    xw_kernel<float><<<(N_NODES + TB-1)/TB, TB, 0, stream>>>(x, e1b2, q1, 32);

    // Layer 1: fused edge-weight GEMM + per-edge matvec + scatter-add
    edge_gemm_kernel<float><<<160, TB, 0, stream>>>(x, h1e, b1s, ei, q1, agg1, 32, 0, npairs);
    node1_kernel<<<(N_NODES + TB-1)/TB, TB, 0, stream>>>(x, root1, bias1, agg1, h1n);

    // Layer 2 (two 32-channel output slabs)
    xw_kernel<_Float16><<<(N_NODES + TB-1)/TB, TB, 0, stream>>>(h1n, e2b2, q2, 64);
    edge_gemm_kernel<_Float16><<<160, TB, 0, stream>>>(h1n, h2e, b2sa, ei, q2, agg2, 64, 0,  npairs);
    edge_gemm_kernel<_Float16><<<160, TB, 0, stream>>>(h1n, h2e, b2sb, ei, q2, agg2, 64, 32, npairs);

    node2_pool_kernel<<<(N_NODES + TB-1)/TB, TB, 0, stream>>>(h1n, root2, bias2, agg2, bat,
                                                              gmax, gsum, gcnt);
    head_kernel<<<1, TB, 0, stream>>>(gmax, gsum, gcnt, h1w, h1b, bn1g, bn1b, bn1m, bn1v,
                                      h2w, h2b, bn2g, bn2b, bn2m, bn2v, h3w, h3b, out);
}
